// ContinuousPositionalBias_6700148981826
// MI455X (gfx1250) — matrix operand order repaired and checked by reading, not compiled, not run on hardware
//
#include <hip/hip_runtime.h>

// ---------------------------------------------------------------------------
// ContinuousPositionalBias for MI455X (gfx1250, wave32, WMMA).
// B=2, Nq=Nk=1024, HD=128, H=8 (fixed by the reference).
//
// Setup kernel: pack W2 (128x128) and W3 (128x8 -> 128x16 pad) once into
//               bf16 B-fragment order in d_ws.
// Main kernel:  grid (4,1024,2), 512 threads = 16 wave32s.
//               Block = one (b,q) x 256-k tile -> M=256 pair rows.
//               Wave w owns rows [16w,16w+16): layer2 = 8 N-tiles x 4 K-chunk
//               v_wmma_f32_16x16x32_bf16 (double-buffered B frags), layer3 =
//               4 more WMMAs against padded W3. h2 aliases h1 in LDS.
// ---------------------------------------------------------------------------

typedef __attribute__((ext_vector_type(16))) __bf16 v16bf;
typedef __attribute__((ext_vector_type(8)))  __bf16 v8bf;
typedef __attribute__((ext_vector_type(8)))  float  v8f;

union AFrag {
    v16bf v;
    int4  q[2];
};

// Dynamic-LDS layout (bytes)
#define OFF_H1   0            // 256*128 bf16 = 65536  (h2 aliases this)
#define OFF_W2P  65536        // packed W2 bf16 = 32768
#define OFF_W3P  98304        // packed W3 bf16 = 4096   (contiguous w/ W2P)
#define OFF_BIAS 102400       // 16*260 f32 = 16640 (head-major, padded)
#define OFF_W1   119040       // 256 f32
#define OFF_B1   120064       // 128 f32
#define OFF_B2   120576       // 128 f32
#define OFF_B3   121088       // 8 f32 (pad)
#define SMEM_BYTES 121152

#define WS_PACK_BYTES 36864   // 32768 (w2p) + 4096 (w3p) in d_ws

// ---------------------------------------------------------------------------
// One-time pack of W2/W3 into bf16 fragment order (B operand 32x16, this kernel's contiguous k order matching the A loads,
// 16-bit: lane = column (lane&15), K = (lane&16) + i, chunks of 32 K).
// Element order: e = tile*2048 + chunk*512 + lane*16 + i.
// ---------------------------------------------------------------------------
__global__ __launch_bounds__(256) void cpb_pack_kernel(
    const float* __restrict__ w2,   // (128,128)
    const float* __restrict__ w3,   // (128,8)
    __bf16* __restrict__ wsp)       // [16384] w2p | [2048] w3p
{
    const int tid = threadIdx.x;
    for (int n = 0; n < 64; ++n) {
        int e     = tid * 64 + n;            // 0..16383
        int i     = e & 15;
        int ln    = (e >> 4) & 31;
        int chunk = (e >> 9) & 3;
        int tile  = e >> 11;
        int col   = tile * 16 + (ln & 15);
        int k     = chunk * 32 + (ln & 16) + i;
        wsp[e] = (__bf16)w2[k * 128 + col];
    }
    for (int n = 0; n < 8; ++n) {
        int e     = tid * 8 + n;             // 0..2047
        int i     = e & 15;
        int ln    = (e >> 4) & 31;
        int chunk = (e >> 9) & 3;
        int col   = ln & 15;
        int k     = chunk * 32 + (ln & 16) + i;
        float v   = (col < 8) ? w3[k * 8 + col] : 0.0f;
        wsp[16384 + e] = (__bf16)v;
    }
}

__global__ __launch_bounds__(512) void cpb_wmma_kernel(
    const float* __restrict__ qcoord,   // (2,1024,2)
    const float* __restrict__ kcoord,   // (2,1024,2)
    const float* __restrict__ w1,       // (2,128)
    const float* __restrict__ b1,       // (128)
    const float* __restrict__ b2,       // (128)
    const float* __restrict__ b3,       // (8)
    const __bf16* __restrict__ wsp,     // packed W2|W3 (36864 B)
    float* __restrict__ out)            // (2,8,1024,1024)
{
    extern __shared__ char smem[];
    char*  h1b     = smem + OFF_H1;           // h1 (and later h2), row=256B
    char*  w2pb    = smem + OFF_W2P;
    char*  w3pb    = smem + OFF_W3P;
    float* biasOut = (float*)(smem + OFF_BIAS);   // [16 heads][260 k]
    float* w1s     = (float*)(smem + OFF_W1);
    float* b1s     = (float*)(smem + OFF_B1);
    float* b2s     = (float*)(smem + OFF_B2);
    float* b3s     = (float*)(smem + OFF_B3);

    const int tid  = threadIdx.x;
    const int lane = tid & 31;
    const int wv   = tid >> 5;          // wave id 0..15 -> M-tile
    const int k0   = blockIdx.x * 256;  // k tile
    const int q    = blockIdx.y;
    const int b    = blockIdx.z;

    // ---------------- Phase A: bulk-copy packed weights, stage params ------
    {
        const int4* src = (const int4*)wsp;
        int4*       dst = (int4*)w2pb;              // W2P|W3P contiguous
        for (int e = tid; e < (WS_PACK_BYTES / 16); e += 512)
            dst[e] = src[e];
    }
    if (tid < 256)               w1s[tid]       = w1[tid];
    else if (tid < 384)          b1s[tid - 256] = b1[tid - 256];
    else                         b2s[tid - 384] = b2[tid - 384];
    if (tid < 8)                 b3s[tid]       = b3[tid];
    __syncthreads();

    // ---------------- Phase B: layer 1 -> h1 (bf16 row-major [256][128]) ---
    {
        const int kl = tid & 255;
        const float2 qc2 = ((const float2*)qcoord)[b * 1024 + q];
        const float2 kc2 = ((const float2*)kcoord)[b * 1024 + k0 + kl];
        const float rx = qc2.x - kc2.x;
        const float ry = qc2.y - kc2.y;
        const int j0 = (tid >> 8) * 64;
        for (int j = j0; j < j0 + 64; j += 8) {
            v8bf pk;
            #pragma unroll
            for (int u = 0; u < 8; ++u) {
                float v = fmaf(rx, w1s[j + u], fmaf(ry, w1s[128 + j + u], b1s[j + u]));
                pk[u] = (__bf16)fmaxf(v, 0.0f);
            }
            *(v8bf*)(h1b + kl * 256 + j * 2) = pk;
        }
    }
    __syncthreads();

    // ---------------- Phase C: layer 2 + layer 3 via WMMA ------------------
    const int sub  = (lane & 16) ? 32 : 0;           // A-layout lane-half skew
    const int rowA = wv * 16 + (lane & 15);
    const int rowb = wv * 16 + ((lane & 16) ? 8 : 0);
    const int colL = lane & 15;

    // A fragments (all 4 K-chunks) for this wave's 16 rows -> registers.
    AFrag aF[4];
    {
        const char* h1row = h1b + rowA * 256;
        #pragma unroll
        for (int c = 0; c < 4; ++c) {
            aF[c].q[0] = *(const int4*)(h1row + c * 64 + sub);
            aF[c].q[1] = *(const int4*)(h1row + c * 64 + 16 + sub);
        }
    }

    // Double-buffered B fragments: tile nt occupies nt*4096 .. +4095 bytes.
    AFrag bF[2][4];
    #pragma unroll
    for (int c = 0; c < 4; ++c) {
        const char* p = w2pb + c * 1024 + lane * 32;
        bF[0][c].q[0] = *(const int4*)p;
        bF[0][c].q[1] = *(const int4*)(p + 16);
    }

    #pragma unroll
    for (int nt = 0; nt < 8; ++nt) {
        if (nt < 7) {                      // prefetch next N-tile's B frags
            #pragma unroll
            for (int c = 0; c < 4; ++c) {
                const char* p = w2pb + (nt + 1) * 4096 + c * 1024 + lane * 32;
                bF[(nt + 1) & 1][c].q[0] = *(const int4*)p;
                bF[(nt + 1) & 1][c].q[1] = *(const int4*)(p + 16);
            }
        }
        v8f acc;
        #pragma unroll
        for (int r = 0; r < 8; ++r) acc[r] = 0.0f;
        #pragma unroll
        for (int c = 0; c < 4; ++c)
            acc = __builtin_amdgcn_wmma_f32_16x16x32_bf16(
                false, aF[c].v, false, bF[nt & 1][c].v, (short)0, acc, false, false);

        // relu(acc + b2) -> bf16 h2 (aliases h1; wave-private rows)
        const float bv = b2s[nt * 16 + colL];
        const int   col = nt * 16 + colL;
        #pragma unroll
        for (int r = 0; r < 8; ++r)
            *(__bf16*)(h1b + (rowb + r) * 256 + col * 2) =
                (__bf16)fmaxf(acc[r] + bv, 0.0f);
    }

    // Layer 3: A = h2 rows of this wave, B = padded W3.
    {
        const char* h2row = h1b + rowA * 256;
        #pragma unroll
        for (int c = 0; c < 4; ++c) {
            aF[c].q[0] = *(const int4*)(h2row + c * 64 + sub);
            aF[c].q[1] = *(const int4*)(h2row + c * 64 + 16 + sub);
        }
        v8f acc;
        #pragma unroll
        for (int r = 0; r < 8; ++r) acc[r] = 0.0f;
        #pragma unroll
        for (int c = 0; c < 4; ++c) {
            AFrag bW;
            const char* p = w3pb + c * 1024 + lane * 32;
            bW.q[0] = *(const int4*)p;
            bW.q[1] = *(const int4*)(p + 16);
            acc = __builtin_amdgcn_wmma_f32_16x16x32_bf16(
                false, aF[c].v, false, bW.v, (short)0, acc, false, false);
        }
        // head-major staging: biasOut[h][k], row stride 260 (bank-spread)
        const float bv = (colL < 8) ? b3s[colL] : 0.0f;
        float4 s0, s1;
        s0.x = acc[0] + bv; s0.y = acc[1] + bv; s0.z = acc[2] + bv; s0.w = acc[3] + bv;
        s1.x = acc[4] + bv; s1.y = acc[5] + bv; s1.z = acc[6] + bv; s1.w = acc[7] + bv;
        float* dst = biasOut + colL * 260 + rowb;
        *(float4*)(dst)     = s0;
        *(float4*)(dst + 4) = s1;
    }
    __syncthreads();

    // ---------------- Phase D: coalesced output (B,H,Nq,Nk) ----------------
    for (int e = tid; e < 2048; e += 512) {
        int h  = e >> 8;
        int kl = e & 255;
        out[(((size_t)(b * 8 + h)) * 1024 + q) * 1024 + (k0 + kl)] =
            biasOut[h * 260 + kl];
    }
}

extern "C" void kernel_launch(void* const* d_in, const int* in_sizes, int n_in,
                              void* d_out, int out_size, void* d_ws, size_t ws_size,
                              hipStream_t stream) {
    (void)in_sizes; (void)n_in; (void)out_size; (void)ws_size;
    const float* qcoord = (const float*)d_in[0];
    const float* kcoord = (const float*)d_in[1];
    const float* w1     = (const float*)d_in[2];
    const float* b1     = (const float*)d_in[3];
    const float* w2     = (const float*)d_in[4];
    const float* b2     = (const float*)d_in[5];
    const float* w3     = (const float*)d_in[6];
    const float* b3     = (const float*)d_in[7];
    float*  out = (float*)d_out;
    __bf16* wsp = (__bf16*)d_ws;

    cpb_pack_kernel<<<dim3(1), dim3(256), 0, stream>>>(w2, w3, wsp);

    dim3 grid(4, 1024, 2);   // (k-tiles of 256, Nq, B)
    dim3 block(512);         // 16 wave32s
    cpb_wmma_kernel<<<grid, block, SMEM_BYTES, stream>>>(
        qcoord, kcoord, w1, b1, b2, b3, wsp, out);
}
